// VoidLinear_9732395892779
// MI455X (gfx1250) — compile-verified, matrix operand order checked by reading, not run on hardware
//
#include <hip/hip_runtime.h>

typedef __attribute__((ext_vector_type(16))) _Float16 v16h;
typedef __attribute__((ext_vector_type(8)))  _Float16 v8h;
typedef __attribute__((ext_vector_type(4)))  _Float16 v4h;
typedef __attribute__((ext_vector_type(8)))  float    v8f;
typedef __attribute__((ext_vector_type(4)))  float    v4f;

#define IN_DIM   4096
#define OUT_DIM  4096
#define M_TOTAL  8192
#define LR_DIM   64

// ---------------------------------------------------------------------------
// Fragment loader: 16x32 fp16 tile, row-major memory (contiguous along K).
// CDNA5 layout (ISA 7.12.2): lanes 0-15 row=lane, K = {0..7, 16..23};
//                            lanes 16-31 row=lane-16, K = {8..15, 24..31}.
// Two 16-byte loads per lane.
// ---------------------------------------------------------------------------
__device__ __forceinline__ v16h load_frag16x32(const _Float16* __restrict__ base,
                                               int stride, int lane) {
    const int row  = lane & 15;
    const int kOff = (lane & 16) ? 8 : 0;
    const _Float16* p = base + (size_t)row * stride + kOff;
    union { v16h v; v8h h[2]; } u;
    u.h[0] = *(const v8h*)(p);        // K = kOff .. kOff+7
    u.h[1] = *(const v8h*)(p + 16);   // K = kOff+16 .. kOff+23
    return u.v;
}

__device__ __forceinline__ v8f wmma_f16(v16h a, v16h b, v8f c) {
    return __builtin_amdgcn_wmma_f32_16x16x32_f16(
        /*neg_a=*/false, a, /*neg_b=*/false, b,
        /*c_mod=*/(short)0, c, /*reuse_a=*/false, /*reuse_b=*/false);
}

// ---------------------------------------------------------------------------
// Kernel 1: m_f16[o][i] = gb[o/8,i/8] + frm[o][i]*gs[o/8,i/8] + sa[o][i]
// (fp16 arithmetic, matching the reference). 8 contiguous fp16 per thread.
// ---------------------------------------------------------------------------
__global__ void build_m_kernel(const _Float16* __restrict__ frm,
                               const float* __restrict__ gs,
                               const float* __restrict__ gb,
                               const float* __restrict__ sa,
                               _Float16* __restrict__ m) {
    const int idx = blockIdx.x * 256 + threadIdx.x;   // 4096*512 threads
    const int o = idx >> 9;                           // output row
    const int t = idx & 511;                          // 8-wide tile column
    const _Float16 s = (_Float16)gs[(o >> 3) * 512 + t];
    const _Float16 b = (_Float16)gb[(o >> 3) * 512 + t];
    const size_t base = (size_t)o * IN_DIM + (size_t)t * 8;
    const v8h f = __builtin_nontemporal_load((const v8h*)(frm + base));
    v8h r;
#pragma unroll
    for (int e = 0; e < 8; ++e)
        r[e] = b + f[e] * s + (_Float16)sa[base + e];
    *(v8h*)(m + base) = r;
}

// ---------------------------------------------------------------------------
// Kernel 2: generic fp32 -> fp16 convert, 4 elements / thread.
// NT load: the fp32 source is streamed once and must not evict the
// L2-resident fp16 working set.
// ---------------------------------------------------------------------------
__global__ void cvt_f32_f16_kernel(const float* __restrict__ in,
                                   _Float16* __restrict__ out, int n4) {
    const int i = blockIdx.x * 256 + threadIdx.x;
    if (i < n4) {
        const v4f v = __builtin_nontemporal_load((const v4f*)in + i);
        v4h h;
#pragma unroll
        for (int e = 0; e < 4; ++e) h[e] = (_Float16)v[e];
        ((v4h*)out)[i] = h;
    }
}

// ---------------------------------------------------------------------------
// Kernel 3: hid = x @ L1.T via WMMA (fp32 accumulate), stored fp16.
// Block = 256 threads = 8 waves; block tile 64(M) x 64(N=LR cols).
// Wave tile 16x32: wm = wave>>1 row group, wn = wave&1 col group.
// x_f16 is read exactly once (67 MB); L1_f16 (512 KB) is cache-resident.
// ---------------------------------------------------------------------------
__global__ __launch_bounds__(256, 1)
void hid_wmma_kernel(const _Float16* __restrict__ xh,   // [8192][4096]
                     const _Float16* __restrict__ l1h,  // [64][4096]
                     _Float16* __restrict__ hid) {      // [8192][64]
    const int lane = threadIdx.x & 31;
    const int w    = threadIdx.x >> 5;
    const int wm   = w >> 1;                 // 0..3
    const int wn   = w & 1;                  // 0..1
    const int rowBase = blockIdx.x * 64 + wm * 16;
    const int colBase = wn * 32;

    v8f acc[2];
#pragma unroll
    for (int j = 0; j < 2; ++j)
#pragma unroll
        for (int e = 0; e < 8; ++e) acc[j][e] = 0.f;

    for (int k = 0; k < IN_DIM; k += 32) {
        const v16h a = load_frag16x32(xh + (size_t)rowBase * IN_DIM + k,
                                      IN_DIM, lane);
#pragma unroll
        for (int j = 0; j < 2; ++j) {
            const v16h b = load_frag16x32(
                l1h + (size_t)(colBase + j * 16) * IN_DIM + k, IN_DIM, lane);
            acc[j] = wmma_f16(a, b, acc[j]);
        }
    }

    const int nlane = lane & 15;
    const int hi8   = (lane & 16) ? 8 : 0;
#pragma unroll
    for (int j = 0; j < 2; ++j) {
        const int col = colBase + j * 16 + nlane;
#pragma unroll
        for (int e = 0; e < 8; ++e)
            hid[(size_t)(rowBase + hi8 + e) * LR_DIM + col] = (_Float16)acc[j][e];
    }
}

// ---------------------------------------------------------------------------
// Kernel 4: main fused GEMM.
// Block = 256 threads = 8 waves; block tile 128(M) x 128(N).
// Wave tile 64x32 = 4x2 accumulators of v_wmma_f32_16x16x32_f16.
// Epilogue: mult/add via 4 WMMAs over K=64 (hid_f16 x l2_f16), + scale/bias,
// out = add + mult * (fp16-rounded baseline). Output stored non-temporally.
// ---------------------------------------------------------------------------
__global__ __launch_bounds__(256, 1)
void void_linear_main(const _Float16* __restrict__ xh,    // [8192][4096]
                      const _Float16* __restrict__ mh,    // [4096][4096]
                      const _Float16* __restrict__ hidh,  // [8192][64]
                      const _Float16* __restrict__ l2h,   // [8192][64]
                      const float* __restrict__ scale,
                      const float* __restrict__ bias,
                      float* __restrict__ out) {
    const int lane = threadIdx.x & 31;
    const int w    = threadIdx.x >> 5;
    const int wm   = w >> 2;          // 0..1
    const int wn   = w & 3;           // 0..3
    const int rowBase = blockIdx.y * 128 + wm * 64;
    const int colBase = blockIdx.x * 128 + wn * 32;

    v8f acc[4][2];
#pragma unroll
    for (int i = 0; i < 4; ++i)
#pragma unroll
        for (int j = 0; j < 2; ++j)
#pragma unroll
            for (int e = 0; e < 8; ++e) acc[i][j][e] = 0.f;

    // ---- main K loop: baseline = x_f16 @ m.T ----
    for (int k = 0; k < IN_DIM; k += 32) {
        v16h a[4], b[2];
#pragma unroll
        for (int i = 0; i < 4; ++i)
            a[i] = load_frag16x32(xh + (size_t)(rowBase + i * 16) * IN_DIM + k,
                                  IN_DIM, lane);
#pragma unroll
        for (int j = 0; j < 2; ++j)
            b[j] = load_frag16x32(mh + (size_t)(colBase + j * 16) * IN_DIM + k,
                                  IN_DIM, lane);
#pragma unroll
        for (int i = 0; i < 4; ++i)
#pragma unroll
            for (int j = 0; j < 2; ++j)
                acc[i][j] = wmma_f16(a[i], b[j], acc[i][j]);
    }

    // ---- epilogue: low-rank mult/add via WMMA over K=64, then fuse ----
    const int nlane = lane & 15;
    const int hi8   = (lane & 16) ? 8 : 0;
#pragma unroll
    for (int i = 0; i < 4; ++i) {
        v16h ha[2];
#pragma unroll
        for (int ks = 0; ks < 2; ++ks)
            ha[ks] = load_frag16x32(
                hidh + (size_t)(rowBase + i * 16) * LR_DIM + ks * 32,
                LR_DIM, lane);
#pragma unroll
        for (int j = 0; j < 2; ++j) {
            const int col0 = colBase + j * 16;
            v8f mc, ac;
#pragma unroll
            for (int e = 0; e < 8; ++e) { mc[e] = 0.f; ac[e] = 0.f; }
#pragma unroll
            for (int ks = 0; ks < 2; ++ks) {
                const v16h mb = load_frag16x32(
                    l2h + (size_t)col0 * LR_DIM + ks * 32, LR_DIM, lane);
                const v16h ab = load_frag16x32(
                    l2h + (size_t)(OUT_DIM + col0) * LR_DIM + ks * 32,
                    LR_DIM, lane);
                mc = wmma_f16(ha[ks], mb, mc);
                ac = wmma_f16(ha[ks], ab, ac);
            }
            const int   col = col0 + nlane;
            const float sc  = scale[col];
            const float bi  = bias[col];
            const int   r0  = rowBase + i * 16 + hi8;
#pragma unroll
            for (int e = 0; e < 8; ++e) {
                const float base = (float)(_Float16)acc[i][j][e]; // fp16 matmul output
                const float mult = mc[e] + sc;
                const float add  = ac[e] + bi;
                __builtin_nontemporal_store(add + mult * base,
                    &out[(size_t)(r0 + e) * OUT_DIM + col]);
            }
        }
    }
}

// ---------------------------------------------------------------------------
// Workspace layout (bytes):
//   m_f16  : [0,            33,554,432)
//   x_f16  : [33,554,432,  100,663,296)
//   hid_f16: [100,663,296, 101,711,872)
//   l2_f16 : [101,711,872, 102,760,448)
//   l1_f16 : [102,760,448, 103,284,736)
// ---------------------------------------------------------------------------
extern "C" void kernel_launch(void* const* d_in, const int* in_sizes, int n_in,
                              void* d_out, int out_size, void* d_ws, size_t ws_size,
                              hipStream_t stream) {
    const float*    x    = (const float*)   d_in[0];
    const _Float16* frm  = (const _Float16*)d_in[1];
    const float*    L1   = (const float*)   d_in[2];
    const float*    L2   = (const float*)   d_in[3];
    const float*    gs   = (const float*)   d_in[4];
    const float*    gb   = (const float*)   d_in[5];
    const float*    sa   = (const float*)   d_in[6];
    const float*    scl  = (const float*)   d_in[7];
    const float*    bia  = (const float*)   d_in[8];
    float*          out  = (float*)d_out;

    char* ws = (char*)d_ws;
    _Float16* m_h   = (_Float16*)(ws);
    _Float16* x_h   = (_Float16*)(ws + 33554432ull);
    _Float16* hid_h = (_Float16*)(ws + 100663296ull);
    _Float16* l2_h  = (_Float16*)(ws + 101711872ull);
    _Float16* l1_h  = (_Float16*)(ws + 102760448ull);

    // 1) modulated weight matrix (fp16)
    build_m_kernel<<<8192, 256, 0, stream>>>(frm, gs, gb, sa, m_h);
    // 2) x -> fp16  (33,554,432 elems / 4 per thread)
    cvt_f32_f16_kernel<<<32768, 256, 0, stream>>>(x, x_h, 8388608);
    // 3) L2 -> fp16 (524,288 elems / 4 per thread)
    cvt_f32_f16_kernel<<<512, 256, 0, stream>>>(L2, l2_h, 131072);
    // 4) L1 -> fp16 (262,144 elems / 4 per thread)
    cvt_f32_f16_kernel<<<256, 256, 0, stream>>>(L1, l1_h, 65536);
    // 5) hid = x @ L1.T via WMMA
    hid_wmma_kernel<<<M_TOTAL / 64, 256, 0, stream>>>(x_h, l1_h, hid_h);
    // 6) fused WMMA GEMM + low-rank epilogue
    dim3 grid(OUT_DIM / 128, M_TOTAL / 128);   // (32, 64)
    void_linear_main<<<grid, 256, 0, stream>>>(x_h, m_h, hid_h, l2_h, scl, bia, out);
}
